// BaseQuantizer_463856467973
// MI455X (gfx1250) — matrix operand order repaired and checked by reading, not compiled, not run on hardware
//
#include <hip/hip_runtime.h>

// ---------------------------------------------------------------------------
// VQ quantizer for MI455X (gfx1250, wave32, WMMA + TDM).
//   x:        [16, 4096, 64] f32   (65536 tokens, D=64)
//   codebook: [64, 1024] f32       (D x K)
//   cluster_frequency: [1024] f32
// Outputs (flat, in tuple order):
//   x_quantized_st [65536*64] | codebook_index (as float) [65536] |
//   inner_loss [1] | new_cluster_frequency [1024]
// ---------------------------------------------------------------------------

typedef __attribute__((ext_vector_type(16))) __bf16    v16bf;
typedef __attribute__((ext_vector_type(8)))  float     v8f;
typedef __attribute__((ext_vector_type(8)))  unsigned  v8u;
typedef __attribute__((ext_vector_type(4)))  unsigned  v4u;
typedef __attribute__((ext_vector_type(8)))  int       v8i;
typedef __attribute__((ext_vector_type(4)))  int       v4i;

#define NTOK    65536
#define DIMD    64
#define KCB     1024
#define NTILES  64             // 1024 codes / 16 per WMMA N-tile
#define ALPHA_F 0.95f

// staged block: 32768 dwords of bf16 B-fragments + 1024 dwords of ||cb||^2
#define STAGE_DW 33792

#if defined(__has_builtin)
#if __has_builtin(__builtin_amdgcn_tensor_load_to_lds) && \
    __has_builtin(__builtin_amdgcn_s_wait_tensorcnt)
#define HAVE_TDM 1
#endif
#endif

// f32 -> bf16 (round-to-nearest-even), pack two into one dword (lo = first)
__device__ __forceinline__ unsigned pack_bf16x2(float lo, float hi) {
  unsigned a = __float_as_uint(lo);
  unsigned b = __float_as_uint(hi);
  a += 0x7FFFu + ((a >> 16) & 1u);
  b += 0x7FFFu + ((b >> 16) & 1u);
  return (a >> 16) | (b & 0xFFFF0000u);
}

// ---------------------------------------------------------------------------
// Prep: build bf16 B-fragments of the codebook in WMMA VGPR layout, compute
// ||cb_k||^2, zero per-call accumulators (histogram + loss).
//
// Fragment layout (ISA 16-bit A/B 16x32): lane l (col N = l&15, half = l>>4),
// dword j (0..7): K pair base = (j<4 ? 2j : 2j+8) + 8*half, elements
// (Kb, Kb+1), plus +32 for the second K-fragment.
// wsfrag[((tile*2 + frag)*32 + lane)*8 + j]
// ---------------------------------------------------------------------------
__global__ __launch_bounds__(256) void vq_prep(
    const float* __restrict__ codebook,
    unsigned* __restrict__ wsfrag, float* __restrict__ wsnorm,
    float* __restrict__ ws_count, float* __restrict__ ws_loss) {
  int bid = blockIdx.x, tid = threadIdx.x;
  if (bid < 128) {
    int gid  = bid * 256 + tid;        // 0..32767
    int t    = gid >> 9;
    int rem  = gid & 511;
    int f    = rem >> 8;
    int rem2 = rem & 255;
    int l    = rem2 >> 3;
    int j    = rem2 & 7;
    int half = l >> 4;
    int n    = t * 16 + (l & 15);
    int kb   = ((j < 4) ? (2 * j) : (2 * j + 8)) + half * 8 + 32 * f;
    float v0 = codebook[(size_t)kb * KCB + n];
    float v1 = codebook[(size_t)(kb + 1) * KCB + n];
    wsfrag[gid] = pack_bf16x2(v0, v1);
  } else {
    int k = (bid - 128) * 256 + tid;   // 0..1023
    float s = 0.f;
    #pragma unroll 8
    for (int d = 0; d < DIMD; ++d) {
      float c = codebook[(size_t)d * KCB + k];
      s = fmaf(c, c, s);
    }
    wsnorm[k]   = s;
    ws_count[k] = 0.f;
    if (k == 0) *ws_loss = 0.f;
  }
}

// ---------------------------------------------------------------------------
// Main: stage fragments+norms into LDS once per workgroup via the Tensor Data
// Mover, then one wave per 16-token M-tile sweeps all 64 codebook tiles with
// v_wmma_f32_16x16x32_bf16 (K=64 -> 2 chained WMMAs) reading B from LDS.
// Tile loop unrolled x2 so two independent acc chains hide the WMMA->VALU
// hazard (bf16 WMMA co-executes with the other tile's argmin VALU).
// ---------------------------------------------------------------------------
__global__ __launch_bounds__(256) void vq_main(
    const float* __restrict__ x, const float* __restrict__ codebook,
    const unsigned* __restrict__ wsfrag,
    float* __restrict__ out_xq, float* __restrict__ out_idx,
    float* __restrict__ ws_count, float* __restrict__ ws_loss) {
  __shared__ __align__(16) unsigned ldsbuf[STAGE_DW];   // 132 KB of 320 KB/WGP

  const int tid  = threadIdx.x;
  const int wave = tid >> 5;
  const int lane = tid & 31;
  const int half = lane >> 4;
  const int col  = lane & 15;
  const int m0   = (blockIdx.x * 8 + wave) * 16;

#if defined(HAVE_TDM)
  if (wave == 0) {
    unsigned lds_off = (unsigned)(size_t)(void*)&ldsbuf[0];
    unsigned long long ga = (unsigned long long)(const void*)wsfrag;
    // D# group 0: count=1 | lds_addr | global_addr[56:0] | type=2
    v4u g0;
    g0[0] = 1u;                                     // valid, 1 descriptor
    g0[1] = lds_off;                                // lds_addr (bytes)
    g0[2] = (unsigned)(ga & 0xFFFFFFFFu);           // global_addr[31:0]
    g0[3] = (unsigned)((ga >> 32) & 0x01FFFFFFu)    // global_addr[56:32]
            | (2u << 30);                           // type = 2 ("image")
    // D# group 1: data_size=4B, 2D tensor 33792 x 1, tile 33792 x 1
    v8i g1;
    g1[0] = (int)(2u << 16);                        // workgroup_mask=0, data_size=2 (4B)
    g1[1] = (int)((STAGE_DW & 0xFFFFu) << 16);      // tensor_dim0[15:0] << 16
    g1[2] = (int)((STAGE_DW >> 16) | (1u << 16));   // tensor_dim0[31:16] | tensor_dim1 lo16=1
    g1[3] = (int)((STAGE_DW & 0xFFFFu) << 16);      // tensor_dim1 hi16=0 | tile_dim0
    g1[4] = 1;                                      // tile_dim1=1, tile_dim2=0
    g1[5] = (int)STAGE_DW;                          // tensor_dim0_stride[31:0]
    g1[6] = 0;                                      // stride0 hi, stride1 lo
    g1[7] = 0;                                      // stride1 hi
    v4i g2 = {0, 0, 0, 0};                          // dims 2/3 unused
    v4i g3 = {0, 0, 0, 0};
    v8i g4 = {0, 0, 0, 0, 0, 0, 0, 0};              // unused trailing group
    __builtin_amdgcn_tensor_load_to_lds(g0, g1, g2, g3, g4, 0);
    __builtin_amdgcn_s_wait_tensorcnt(0);
  }
#else
  for (int i = tid; i < STAGE_DW / 4; i += 256)
    ((uint4*)ldsbuf)[i] = ((const uint4*)wsfrag)[i];
#endif
  __syncthreads();

  // ---- A fragments: x rows m0..m0+15, bf16, ISA 16-bit A layout ----
  const float4* xr = (const float4*)(x + (size_t)(m0 + col) * DIMD);
  v16bf afrag[2];
  #pragma unroll
  for (int f = 0; f < 2; ++f) {
    const int base = 8 * half + 32 * f;            // runs base..+7, base+16..+23
    float4 A0 = xr[(base >> 2) + 0];
    float4 A1 = xr[(base >> 2) + 1];
    float4 B0 = xr[((base + 16) >> 2) + 0];
    float4 B1 = xr[((base + 16) >> 2) + 1];
    v8u u;
    u[0] = pack_bf16x2(A0.x, A0.y); u[1] = pack_bf16x2(A0.z, A0.w);
    u[2] = pack_bf16x2(A1.x, A1.y); u[3] = pack_bf16x2(A1.z, A1.w);
    u[4] = pack_bf16x2(B0.x, B0.y); u[5] = pack_bf16x2(B0.z, B0.w);
    u[6] = pack_bf16x2(B1.x, B1.y); u[7] = pack_bf16x2(B1.z, B1.w);
    afrag[f] = __builtin_bit_cast(v16bf, u);
  }

  float cmin[8];
  int   cidx[8];
  #pragma unroll
  for (int r = 0; r < 8; ++r) { cmin[r] = 3.4e38f; cidx[r] = 0; }

  const float* ldsnorm = (const float*)&ldsbuf[32768];

  // ---- sweep all codebook tiles (B from LDS); unroll 2 for latency hiding --
  #pragma unroll 2
  for (int t = 0; t < NTILES; ++t) {
    const uint4* bp = (const uint4*)&ldsbuf[((t * 2) * 32 + lane) * 8];
    uint4 q0 = bp[0], q1 = bp[1];        // frag 0 (K=0..31)
    uint4 q2 = bp[64], q3 = bp[65];      // frag 1 (K=32..63): +32 lanes * 8 dw
    v8u u0 = {q0.x, q0.y, q0.z, q0.w, q1.x, q1.y, q1.z, q1.w};
    v8u u1 = {q2.x, q2.y, q2.z, q2.w, q3.x, q3.y, q3.z, q3.w};
    v16bf b0 = __builtin_bit_cast(v16bf, u0);
    v16bf b1 = __builtin_bit_cast(v16bf, u1);

    v8f acc = {};
    acc = __builtin_amdgcn_wmma_f32_16x16x32_bf16(false, afrag[0], false, b0,
                                                  (short)0, acc, false, false);
    acc = __builtin_amdgcn_wmma_f32_16x16x32_bf16(false, afrag[1], false, b1,
                                                  (short)0, acc, false, false);

    const float cbn = ldsnorm[t * 16 + col];
    const int   n   = t * 16 + col;
    #pragma unroll
    for (int r = 0; r < 8; ++r) {
      float s = fmaf(-2.f, acc[r], cbn);  // ||cb||^2 - 2 x.cb (row-const ||x||^2 dropped)
      if (s < cmin[r]) { cmin[r] = s; cidx[r] = n; }
    }
  }

  // ---- cross-lane argmin within each 16-lane half (rows stay in-half) ----
  #pragma unroll
  for (int r = 0; r < 8; ++r) {
    float mv = cmin[r];
    int   iv = cidx[r];
    #pragma unroll
    for (int off = 1; off < 16; off <<= 1) {
      float mo = __shfl_xor(mv, off, 32);
      int   io = __shfl_xor(iv, off, 32);
      if (mo < mv || (mo == mv && io < iv)) { mv = mo; iv = io; }
    }
    cmin[r] = mv; cidx[r] = iv;
  }

  // ---- fused gather, straight-through output, loss, histogram, indices ----
  float lsum = 0.f;
  #pragma unroll
  for (int r = 0; r < 8; ++r) {
    const int m   = m0 + 8 * half + r;
    const int idx = cidx[r];
    #pragma unroll
    for (int q = 0; q < 4; ++q) {
      const int d  = col + 16 * q;
      float xq = codebook[(size_t)d * KCB + idx];
      float xv = x[(size_t)m * DIMD + d];
      out_xq[(size_t)m * DIMD + d] = xq;   // x + sg(x_q - x) == x_q numerically
      float df = xq - xv;
      lsum = fmaf(df, df, lsum);
    }
    if (col == 0) {
      out_idx[m] = (float)idx;
      atomicAdd(&ws_count[idx], 1.0f);
    }
  }
  #pragma unroll
  for (int off = 1; off < 32; off <<= 1) lsum += __shfl_xor(lsum, off, 32);
  if (lane == 0) atomicAdd(ws_loss, lsum);
}

// ---------------------------------------------------------------------------
// Finalize: EMA of cluster frequency + loss normalization (x2: commit+codebook)
// ---------------------------------------------------------------------------
__global__ __launch_bounds__(256) void vq_fin(
    const float* __restrict__ cf, const float* __restrict__ ws_count,
    const float* __restrict__ ws_loss,
    float* __restrict__ out_loss, float* __restrict__ out_freq) {
  int k = blockIdx.x * 256 + threadIdx.x;
  if (k < KCB) out_freq[k] = ALPHA_F * cf[k] + (1.0f - ALPHA_F) * ws_count[k];
  if (k == 0)  *out_loss = 2.0f * (*ws_loss) / (float)(NTOK * DIMD);
}

extern "C" void kernel_launch(void* const* d_in, const int* in_sizes, int n_in,
                              void* d_out, int out_size, void* d_ws, size_t ws_size,
                              hipStream_t stream) {
  const float* x        = (const float*)d_in[0];
  const float* codebook = (const float*)d_in[1];
  const float* cf       = (const float*)d_in[2];

  float* out      = (float*)d_out;
  float* out_xq   = out;                              // 4194304
  float* out_idx  = out + (size_t)NTOK * DIMD;        // 65536
  float* out_loss = out_idx + NTOK;                   // 1
  float* out_freq = out_loss + 1;                     // 1024

  unsigned* wsfrag  = (unsigned*)d_ws;                          // 131072 B
  float*    wsnorm  = (float*)((char*)d_ws + 131072);           // 4096 B (contiguous w/ frags)
  float*    wscount = (float*)((char*)d_ws + 135168);           // 4096 B
  float*    wsloss  = (float*)((char*)d_ws + 139264);           // 4 B

  vq_prep<<<132, 256, 0, stream>>>(codebook, wsfrag, wsnorm, wscount, wsloss);
  vq_main<<<512, 256, 0, stream>>>(x, codebook, wsfrag,
                                   out_xq, out_idx, wscount, wsloss);
  vq_fin<<<4, 256, 0, stream>>>(cf, wscount, wsloss, out_loss, out_freq);
}
